// EMDLayer_1743756722663
// MI455X (gfx1250) — matrix operand order repaired and checked by reading, not compiled, not run on hardware
//
#include <hip/hip_runtime.h>

typedef __attribute__((ext_vector_type(16))) _Float16 v16h;
typedef __attribute__((ext_vector_type(2)))  _Float16 h2;
typedef __attribute__((ext_vector_type(8)))  float    v8f;

#define C_DIM   512
#define N_NODE  64
#define QN      256
#define WN      32
#define EPS     0.05f
#define ITERS   100
#define TEMP    12.5f

// ---------------------------------------------------------------------------
// Kernel A: spatial mean pool  out[b][c] = mean_n in[b][c][n]   (n contiguous)
// ---------------------------------------------------------------------------
__global__ void pool_mean_kernel(const float* __restrict__ in,
                                 float* __restrict__ out, int total /*B*C*/) {
    int id = blockIdx.x * blockDim.x + threadIdx.x;
    if (id >= total) return;
    const float4* p = (const float4*)(in + (size_t)id * N_NODE);
    float s = 0.f;
#pragma unroll
    for (int i = 0; i < 16; ++i) { float4 v = p[i]; s += v.x + v.y + v.z + v.w; }
    out[id] = s * (1.0f / 64.0f);
}

// ---------------------------------------------------------------------------
// Kernel B: per-node channel-center + L2 normalize -> f16 [b*64+n][c]
// ---------------------------------------------------------------------------
__global__ void normalize_kernel(const float* __restrict__ in,
                                 _Float16* __restrict__ out) {
    int node = blockIdx.x;            // b*64 + n
    int b = node >> 6, n = node & 63;
    int t = threadIdx.x;              // 0..255
    const float* base = in + (size_t)b * C_DIM * N_NODE + n;
    float x0 = base[(size_t)t * N_NODE];
    float x1 = base[(size_t)(t + 256) * N_NODE];
    __shared__ float red[256];
    red[t] = x0 + x1;
    __syncthreads();
    for (int off = 128; off > 0; off >>= 1) {
        if (t < off) red[t] += red[t + off];
        __syncthreads();
    }
    float m = red[0] * (1.0f / 512.0f);
    __syncthreads();
    float y0 = x0 - m, y1 = x1 - m;
    red[t] = y0 * y0 + y1 * y1;
    __syncthreads();
    for (int off = 128; off > 0; off >>= 1) {
        if (t < off) red[t] += red[t + off];
        __syncthreads();
    }
    float inv = 1.0f / fmaxf(sqrtf(red[0]), 1e-8f);
    _Float16* o = out + (size_t)node * C_DIM;
    o[t]       = (_Float16)(y0 * inv);
    o[t + 256] = (_Float16)(y1 * inv);
}

// ---------------------------------------------------------------------------
// Kernel C: Sinkhorn marginals a,b per (q,w); one block per pair, 256 threads
// ---------------------------------------------------------------------------
__global__ void marginals_kernel(const float* __restrict__ query,
                                 const float* __restrict__ support,
                                 const float* __restrict__ Qp,
                                 const float* __restrict__ Bp,
                                 float* __restrict__ aMar,
                                 float* __restrict__ bMar) {
    int w = blockIdx.x, q = blockIdx.y, t = threadIdx.x;
    __shared__ float bpRow[C_DIM], qpRow[C_DIM], red[256], wvec[N_NODE];
    bpRow[t] = Bp[w * C_DIM + t];  bpRow[t + 256] = Bp[w * C_DIM + t + 256];
    qpRow[t] = Qp[q * C_DIM + t];  qpRow[t + 256] = Qp[q * C_DIM + t + 256];
    __syncthreads();
    int n = t & 63, cq = t >> 6;
#pragma unroll
    for (int pass = 0; pass < 2; ++pass) {
        const float* dat = pass ? (support + (size_t)w * C_DIM * N_NODE)
                                : (query   + (size_t)q * C_DIM * N_NODE);
        const float* vec = pass ? qpRow : bpRow;
        float s = 0.f;
        for (int c = cq * 128; c < cq * 128 + 128; ++c)
            s += dat[(size_t)c * N_NODE + n] * vec[c];   // coalesced over n
        red[t] = s;
        __syncthreads();
        if (cq == 0) {
            float tot = red[n] + red[n + 64] + red[n + 128] + red[n + 192];
            wvec[n] = fmaxf(tot, 0.f) + 0.001f + 1e-5f;
        }
        __syncthreads();
        if (t == 0) {
            float ss = 0.f;
            for (int i = 0; i < N_NODE; ++i) ss += wvec[i];
            red[0] = ss;
        }
        __syncthreads();
        float scale = 64.0f / red[0];
        float* dst = pass ? bMar : aMar;
        if (t < 64) dst[((size_t)q * WN + w) * N_NODE + t] = wvec[t] * scale;
        __syncthreads();
    }
}

// ---------------------------------------------------------------------------
// Kernel D: fused WMMA cosine-sim (64x64, K=512) + multiplicative Sinkhorn.
// One block per (q,w); 8 wave32 waves. GEMM staging via double-buffered
// global_load_async_to_lds_b128 (ASYNCcnt); Sinkhorn kernel matrix lives in
// VGPRs (16 K-row + 16 K^T-col values per thread), only u/v touch LDS.
// ---------------------------------------------------------------------------
#define SA_OFF   0                      // A ping-pong: 2 * 4096 B
#define SB_OFF   (2 * 4096)             // B ping-pong: 2 * 4096 B
#define SIM_OFF  (4 * 4096)             // 64*64 f32 = 16 KB
#define SU_OFF   (SIM_OFF + 16384)
#define SV_OFF   (SU_OFF + 256)
#define SAM_OFF  (SV_OFF + 256)
#define SBM_OFF  (SAM_OFF + 256)
#define RED_OFF  (SBM_OFF + 256)
#define LDS_TOT  (RED_OFF + 1024)       // ~34 KB -> 8 blocks resident per WGP

__global__ __launch_bounds__(256) void emd_kernel(
        const _Float16* __restrict__ Qn, const _Float16* __restrict__ Sn,
        const float* __restrict__ aMar, const float* __restrict__ bMar,
        float* __restrict__ out) {
    int w = blockIdx.x, q = blockIdx.y;
    int t = threadIdx.x;
    int lane = t & 31, wave = t >> 5;
    int hi = lane >> 4, l15 = lane & 15;

    __shared__ __align__(16) char smem[LDS_TOT];
    float* sSim = (float*)(smem + SIM_OFF);
    float* su   = (float*)(smem + SU_OFF);
    float* sv   = (float*)(smem + SV_OFF);
    float* sa   = (float*)(smem + SAM_OFF);
    float* sb   = (float*)(smem + SBM_OFF);
    float* red  = (float*)(smem + RED_OFF);

    const _Float16* Qbase = Qn + (size_t)q * 64 * C_DIM;
    const _Float16* Sbase = Sn + (size_t)w * 64 * C_DIM;

    int tr = wave >> 1;                    // output tile row  -> rows tr*16
    int tc = wave & 1;                     // output col half  -> cols tc*32
    v8f acc0 = {}, acc1 = {};

    int srow = t >> 2, sseg = t & 3;       // staging: 16 B / thread / matrix

    // issue async copy of one 64x32 f16 k-slice of A and B into buffer `buf`
    auto issue = [&](int buf, int kof) {
        unsigned la = (unsigned)(SA_OFF + buf * 4096 + srow * 64 + sseg * 16);
        unsigned lb = (unsigned)(SB_OFF + buf * 4096 + srow * 64 + sseg * 16);
        const char* ga = (const char*)(Qbase + (size_t)srow * C_DIM + kof) + sseg * 16;
        const char* gb = (const char*)(Sbase + (size_t)srow * C_DIM + kof) + sseg * 16;
        asm volatile("global_load_async_to_lds_b128 %0, %1, off"
                     :: "v"(la), "v"(ga) : "memory");
        asm volatile("global_load_async_to_lds_b128 %0, %1, off"
                     :: "v"(lb), "v"(gb) : "memory");
    };

    issue(0, 0);                           // prologue: k-step 0 into buf 0
    for (int kk = 0; kk < 16; ++kk) {
        int cur = kk & 1;
        if (kk < 15) {
            issue(cur ^ 1, (kk + 1) * 32); // prefetch next k-step
            // async loads retire in order: <=2 outstanding => current pair done
            asm volatile("s_wait_asynccnt 0x2" ::: "memory");
        } else {
            asm volatile("s_wait_asynccnt 0x0" ::: "memory");
        }
        __syncthreads();

        const _Float16* bufA = (const _Float16*)(smem + SA_OFF + cur * 4096);
        const _Float16* bufB = (const _Float16*)(smem + SB_OFF + cur * 4096);

        // A fragment (16x32 f16, ISA 7.12.2 wave32 layout)
        v16h af, b0, b1;
        int arow = tr * 16 + l15;
#pragma unroll
        for (int v = 0; v < 8; ++v) {
            int kb = (v < 4 ? 2 * v : 8 + 2 * v) + (hi ? 8 : 0);
            h2 pr = *(const h2*)(&bufA[arow * 32 + kb]);
            af[2 * v] = pr[0]; af[2 * v + 1] = pr[1];
        }
        // B fragments (32x16 f16): bufB holds B[k][n] as [n][k]
        int n0 = tc * 32 + l15, n1 = n0 + 16;
#pragma unroll
        for (int v = 0; v < 8; ++v) {
            int kb = (v < 4 ? 2 * v : 8 + 2 * v) + (hi ? 8 : 0);
            h2 p0 = *(const h2*)(&bufB[n0 * 32 + kb]);
            h2 p1 = *(const h2*)(&bufB[n1 * 32 + kb]);
            b0[2 * v] = p0[0]; b0[2 * v + 1] = p0[1];
            b1[2 * v] = p1[0]; b1[2 * v + 1] = p1[1];
        }
        acc0 = __builtin_amdgcn_wmma_f32_16x16x32_f16(false, af, false, b0,
                                                      (short)0, acc0, false, false);
        acc1 = __builtin_amdgcn_wmma_f32_16x16x32_f16(false, af, false, b1,
                                                      (short)0, acc1, false, false);
        __syncthreads();                  // reads done before buffer reuse
    }
    // spill accumulators: C/D layout — lane n=l15, VGPR r -> M = r + hi*8
#pragma unroll
    for (int r = 0; r < 8; ++r) {
        int m = tr * 16 + r + hi * 8;
        sSim[m * 64 + tc * 32 + l15]      = acc0[r];
        sSim[m * 64 + tc * 32 + 16 + l15] = acc1[r];
    }
    if (t < 64) {
        sa[t] = aMar[((size_t)q * WN + w) * 64 + t];
        sb[t] = bMar[((size_t)q * WN + w) * 64 + t];
        sv[t] = 1.0f;                     // v0 = exp(0)
    }
    __syncthreads();

    // Kernel matrix in registers: K = exp(sim/eps); the constant exp(-1/eps)
    // is factored out and cancels exactly in the transport plan.
    int row = t >> 2, qd = t & 3;         // thread owns row `row`, quarter qd
    float kR[16], kT[16];
#pragma unroll
    for (int j = 0; j < 16; ++j)
        kR[j] = __expf(sSim[row * 64 + qd * 16 + j] * (1.0f / EPS));
#pragma unroll
    for (int i = 0; i < 16; ++i)
        kT[i] = __expf(sSim[(qd * 16 + i) * 64 + row] * (1.0f / EPS));

    // multiplicative Sinkhorn: u = a/(K v); v = b/(K^T u)
    for (int it = 0; it < ITERS; ++it) {
        float p = 0.f;
#pragma unroll
        for (int j = 0; j < 16; ++j) p += kR[j] * sv[qd * 16 + j];
        p += __shfl_xor(p, 1);
        p += __shfl_xor(p, 2);
        if (qd == 0) su[row] = sa[row] / p;
        __syncthreads();
        p = 0.f;
#pragma unroll
        for (int i = 0; i < 16; ++i) p += kT[i] * su[qd * 16 + i];
        p += __shfl_xor(p, 1);
        p += __shfl_xor(p, 2);
        if (qd == 0) sv[row] = sb[row] / p;
        __syncthreads();
    }

    // logits[q,w] = sum_ij sim * (K u v) * T/N
    float p = 0.f;
#pragma unroll
    for (int j = 0; j < 16; ++j)
        p += sSim[row * 64 + qd * 16 + j] * kR[j] * sv[qd * 16 + j];
    p *= su[row];
    red[t] = p;
    __syncthreads();
    for (int off = 128; off > 0; off >>= 1) {
        if (t < off) red[t] += red[t + off];
        __syncthreads();
    }
    if (t == 0) out[q * WN + w] = red[0] * (TEMP / 64.0f);
}

// ---------------------------------------------------------------------------
extern "C" void kernel_launch(void* const* d_in, const int* in_sizes, int n_in,
                              void* d_out, int out_size, void* d_ws, size_t ws_size,
                              hipStream_t stream) {
    const float* support = (const float*)d_in[0];   // [32,512,8,8]
    const float* query   = (const float*)d_in[1];   // [256,512,8,8]
    float* out = (float*)d_out;                     // [256,32]

    char* ws = (char*)d_ws;
    float*    Qp = (float*)(ws + 0);                // 256*512 f32  (512 KB)
    float*    Bp = (float*)(ws + 524288);           //  32*512 f32  ( 64 KB)
    _Float16* QnP = (_Float16*)(ws + 589824);       // 256*64*512 f16 (16 MB)
    _Float16* SnP = (_Float16*)(ws + 17367040);     //  32*64*512 f16 ( 2 MB)
    float*    aM = (float*)(ws + 19464192);         // 8192*64 f32  ( 2 MB)
    float*    bM = (float*)(ws + 21561344);         // 8192*64 f32  ( 2 MB)

    pool_mean_kernel<<<512, 256, 0, stream>>>(query,   Qp, QN * C_DIM);
    pool_mean_kernel<<<64,  256, 0, stream>>>(support, Bp, WN * C_DIM);
    normalize_kernel<<<QN * N_NODE, 256, 0, stream>>>(query,   QnP);
    normalize_kernel<<<WN * N_NODE, 256, 0, stream>>>(support, SnP);
    marginals_kernel<<<dim3(WN, QN), 256, 0, stream>>>(query, support, Qp, Bp, aM, bM);
    emd_kernel<<<dim3(WN, QN), 256, 0, stream>>>(QnP, SnP, aM, bM, out);
}
